// CausalSTFTLayer_33139967656615
// MI455X (gfx1250) — matrix operand order repaired and checked by reading, not compiled, not run on hardware
//
#include <hip/hip_runtime.h>

#define WIN   64
#define HID   768
#define BATCH 2
#define SEQ   1024
#define ROWS  (BATCH * SEQ)   // 2048
#define K2STEPS (HID / 32)    // 24

typedef __attribute__((ext_vector_type(16))) __bf16       v16bf;
typedef __attribute__((ext_vector_type(8)))  float        v8f;
typedef __attribute__((ext_vector_type(4)))  unsigned int v4u;
typedef __attribute__((ext_vector_type(8)))  int          v8i;
typedef __attribute__((ext_vector_type(4)))  int          v4i;

#if __has_include(<hip/amd_detail/amd_gfx1250_TDM.h>)
#define TDM_6ARG 1
#endif

static __device__ __forceinline__ v8f wmma_bf16(v16bf a, v16bf b, v8f c) {
  return __builtin_amdgcn_wmma_f32_16x16x32_bf16(false, a, false, b,
                                                 (short)0, c, false, false);
}

// ---------------------------------------------------------------------------
// TDM: DMA a 2D tile (tile_d1 rows x tile_d0 elements, 2-byte elements) from
// a row-major bf16 tensor (row stride = stride0 elements) into LDS.
// D# packing per CDNA5 ISA ch.8 (group0: count/lds/global/type; group1: dims).
// ---------------------------------------------------------------------------
static __device__ __forceinline__ void tdm_load_2d_bf16(
    const __bf16* lds_dst, const __bf16* gsrc,
    unsigned tensor_d0, unsigned tensor_d1,
    unsigned tile_d0, unsigned tile_d1, unsigned stride0) {
  const unsigned long long ga = (unsigned long long)(uintptr_t)gsrc;
  v4u g0;
  g0[0] = 1u;                                   // count=1, user D#, no gather
  g0[1] = (unsigned)(uintptr_t)lds_dst;         // lds_addr (LDS byte offset)
  g0[2] = (unsigned)(ga & 0xffffffffu);         // global_addr[31:0]
  g0[3] = (unsigned)((ga >> 32) & 0x01ffffffu)  // global_addr[56:32]
        | (2u << 30);                           // type = 2 ("image")
  v8i g1;
  g1[0] = (int)(1u << 16);                      // data_size=1 (2B); mask=0
  g1[1] = (int)((tensor_d0 & 0xffffu) << 16);               // dim0[15:0]
  g1[2] = (int)(((tensor_d0 >> 16) & 0xffffu)
        | ((tensor_d1 & 0xffffu) << 16));                   // dim0[31:16]|dim1[15:0]
  g1[3] = (int)(((tensor_d1 >> 16) & 0xffffu)
        | ((tile_d0 & 0xffffu) << 16));                     // dim1[31:16]|tile0
  g1[4] = (int)(tile_d1 & 0xffffu);                         // tile1; tile2=0
  g1[5] = (int)stride0;                                     // dim0_stride[31:0]
  g1[6] = 0;
  g1[7] = 0;
  const v4i z = {0, 0, 0, 0};
#if defined(TDM_6ARG)
  const v8i z8 = {0, 0, 0, 0, 0, 0, 0, 0};
  __builtin_amdgcn_tensor_load_to_lds(g0, g1, z, z, z8, 0);
#else
  __builtin_amdgcn_tensor_load_to_lds(g0, g1, z, z, 0);
#endif
}

// ---------------------------------------------------------------------------
// Kernel 1: Xc[t,k2] = sum_h x[t,h] cos(2*pi*k2*h/768); Xs with sin.
// WMMA GEMM 2048x768x768 with in-register twiddle generation.
// ---------------------------------------------------------------------------
__global__ __launch_bounds__(128) void k_xtransform(const float* __restrict__ x,
                                                    __bf16* __restrict__ Xc,
                                                    __bf16* __restrict__ Xs) {
  const int lane = threadIdx.x & 31;
  const int wave = threadIdx.x >> 5;
  const int rowL = lane & 15;
  const int hi   = lane >> 4;
  const int t    = blockIdx.x * 16 + rowL;
  const int col  = blockIdx.y * 64 + wave * 16 + rowL;

  const int kbaseA = hi ? 8 : 0;
  const int kbaseB = hi ? 8 : 0;
  const float step = 6.283185307179586f / (float)HID;

  v8f accC = {};
  v8f accS = {};

  for (int h0 = 0; h0 < HID; h0 += 32) {
    v16bf a;
    const float* pa = x + (size_t)t * HID + h0 + kbaseA;
#pragma unroll
    for (int j = 0; j < 8; ++j) a[j] = (__bf16)pa[j];
#pragma unroll
    for (int j = 0; j < 8; ++j) a[8 + j] = (__bf16)pa[16 + j];

    v16bf bc, bs;
#pragma unroll
    for (int j = 0; j < 16; ++j) {
      const int h = h0 + kbaseB + (j < 8 ? j : j + 8);
      const int m = (h * col) % HID;
      float sv, cv;
      __sincosf((float)m * step, &sv, &cv);
      bc[j] = (__bf16)cv;
      bs[j] = (__bf16)sv;
    }
    accC = wmma_bf16(a, bc, accC);
    accS = wmma_bf16(a, bs, accS);
  }

#pragma unroll
  for (int v = 0; v < 8; ++v) {
    const int r = blockIdx.x * 16 + v + (hi ? 8 : 0);
    Xc[(size_t)r * HID + col] = (__bf16)accC[v];
    Xs[(size_t)r * HID + col] = (__bf16)accS[v];
  }
}

// ---------------------------------------------------------------------------
// Kernel 2: fold the window-axis DFT into the weights.
//   Qc[w][k2][h'] =  sum_k1 proj_w[h',k1*768+k2] * cos(2*pi*k1*w/64)
//   Qs[w][k2][h'] = -sum_k1 proj_w[h',k1*768+k2] * sin(2*pi*k1*w/64)
// ---------------------------------------------------------------------------
__global__ __launch_bounds__(256) void k_qbuild(const float* __restrict__ pw,
                                                __bf16* __restrict__ Qc,
                                                __bf16* __restrict__ Qs) {
  __shared__ float cw[WIN * WIN];
  __shared__ float sw[WIN * WIN];
  const int tid = threadIdx.x;

  for (int e = tid; e < WIN * WIN; e += 256) {
    const int k1 = e >> 6, w = e & 63;
    const int m = (k1 * w) & 63;
    float sv, cv;
    __sincosf((float)m * (6.283185307179586f / 64.0f), &sv, &cv);
    cw[e] = cv;
    sw[e] = sv;
  }
  __syncthreads();

  const int gid = blockIdx.x * 256 + tid;
  const int hq  = gid / HID;
  const int k2  = gid % HID;

  float v[WIN];
#pragma unroll 4
  for (int k1 = 0; k1 < WIN; ++k1)
    v[k1] = pw[(size_t)hq * (WIN * HID) + k1 * HID + k2];

  for (int w = 0; w < WIN; ++w) {
    float qc = 0.f, qs = 0.f;
#pragma unroll 8
    for (int k1 = 0; k1 < WIN; ++k1) {
      qc += v[k1] * cw[k1 * WIN + w];
      qs += v[k1] * sw[k1 * WIN + w];
    }
    const size_t o = ((size_t)w * HID + k2) * HID + hq;
    Qc[o] = (__bf16)qc;
    Qs[o] = (__bf16)(-qs);
  }
}

// ---------------------------------------------------------------------------
// Kernel 3: main causal-conv GEMM (M=2048, N=768, K=2*64*768).
// Block = 256 threads = 8 waves; block tile 128(t) x 64(h').
// Wave grid 4x2, each wave a 32x32 macro-tile (4 accumulators, 8 WMMA/K-step).
// Q tiles are DMA'd by the TDM into double-buffered LDS: wave 0 issues the
// next tile pair, then S_WAIT_TENSORCNT<=2 (tensor ops of one wave complete
// in order) so the current buffer is ready while the next pair stays in
// flight behind the WMMAs.
// ---------------------------------------------------------------------------
__global__ __launch_bounds__(256) void k_main(const __bf16* __restrict__ Xc,
                                              const __bf16* __restrict__ Xs,
                                              const __bf16* __restrict__ Qc,
                                              const __bf16* __restrict__ Qs,
                                              const float* __restrict__ bias,
                                              float* __restrict__ out) {
  __shared__ __bf16 Bc[2][32 * 64];   // ping-pong [k(32)][n(64)]
  __shared__ __bf16 Bs[2][32 * 64];

  const int tid  = threadIdx.x;
  const int lane = tid & 31;
  const int wave = tid >> 5;
  const int wr   = wave & 3;    // 4 row groups of 32 t
  const int wc   = wave >> 2;   // 2 col groups of 32 h'
  const int hi   = lane >> 4;
  const int rowL = lane & 15;

  const int t0 = blockIdx.x * 128;
  const int n0 = blockIdx.y * 64;

  const int kbaseA = hi ? 8 : 0;
  const int kbaseB = hi ? 8 : 0;

  v8f acc00 = {}, acc01 = {}, acc10 = {}, acc11 = {};

  auto issue = [&](int it, int buf) {
    const int iw   = it / K2STEPS;
    const int ik2c = (it - iw * K2STEPS) * 32;
    const size_t row = (size_t)iw * HID + ik2c;
    tdm_load_2d_bf16(Bc[buf], Qc + row * HID + n0, HID, WIN * HID, 64, 32, HID);
    tdm_load_2d_bf16(Bs[buf], Qs + row * HID + n0, HID, WIN * HID, 64, 32, HID);
  };

  const int NIT = WIN * K2STEPS;
  if (wave == 0) issue(0, 0);       // prologue: fill buffer 0

  int w = 0, k2c = 0;
  for (int it = 0; it < NIT; ++it) {
    const int buf = it & 1;

    if (wave == 0) {
      if (it + 1 < NIT) {
        issue(it + 1, buf ^ 1);                  // next pair into other buffer
        __builtin_amdgcn_s_wait_tensorcnt(2);    // current pair done, next in flight
      } else {
        __builtin_amdgcn_s_wait_tensorcnt(0);
      }
    }
    __syncthreads();                 // LDS tile visible to all waves

    const int shift = w - (WIN - 1);

    // ---- A fragments: two 16-row groups, cos & sin streams ----
    v16bf ac[2], as[2];
#pragma unroll
    for (int rg = 0; rg < 2; ++rg) {
      const int t  = t0 + wr * 32 + rg * 16 + rowL;
      const int s  = t & (SEQ - 1);
      const bool ok = (s + shift) >= 0;          // causal zero padding
      v16bf tc = {}, ts = {};
      if (ok) {
        const __bf16* pc = Xc + (size_t)(t + shift) * HID + k2c + kbaseA;
        const __bf16* ps = Xs + (size_t)(t + shift) * HID + k2c + kbaseA;
#pragma unroll
        for (int j = 0; j < 8; ++j) { tc[j] = pc[j];          ts[j] = ps[j]; }
#pragma unroll
        for (int j = 0; j < 8; ++j) { tc[8 + j] = pc[16 + j]; ts[8 + j] = ps[16 + j]; }
        if (k2c + 32 < HID) __builtin_prefetch(pc + 32, 0, 0);
      }
      ac[rg] = tc;
      as[rg] = ts;
    }

    // ---- B fragments from LDS: two 16-col groups ----
    v16bf bc[2], bs[2];
#pragma unroll
    for (int cg = 0; cg < 2; ++cg) {
      const int col = wc * 32 + cg * 16 + rowL;
#pragma unroll
      for (int j = 0; j < 16; ++j) {
        bc[cg][j] = Bc[buf][(kbaseB + (j < 8 ? j : j + 8)) * 64 + col];
        bs[cg][j] = Bs[buf][(kbaseB + (j < 8 ? j : j + 8)) * 64 + col];
      }
    }

    // ---- 8 WMMAs: 2x2 tiles x {cos, sin} ----
    acc00 = wmma_bf16(ac[0], bc[0], acc00);
    acc00 = wmma_bf16(as[0], bs[0], acc00);
    acc01 = wmma_bf16(ac[0], bc[1], acc01);
    acc01 = wmma_bf16(as[0], bs[1], acc01);
    acc10 = wmma_bf16(ac[1], bc[0], acc10);
    acc10 = wmma_bf16(as[1], bs[0], acc10);
    acc11 = wmma_bf16(ac[1], bc[1], acc11);
    acc11 = wmma_bf16(as[1], bs[1], acc11);

    __syncthreads();                 // all reads done before buffer refill

    k2c += 32;
    if (k2c == HID) { k2c = 0; ++w; }
  }

  // ---- epilogue: out = acc + bias ----
#pragma unroll
  for (int cg = 0; cg < 2; ++cg) {
    const int hOut = n0 + wc * 32 + cg * 16 + rowL;
    const float bv = bias[hOut];
    const v8f a0 = cg ? acc01 : acc00;
    const v8f a1 = cg ? acc11 : acc10;
#pragma unroll
    for (int v = 0; v < 8; ++v) {
      const int r0 = t0 + wr * 32 + v + (hi ? 8 : 0);
      out[(size_t)r0 * HID + hOut] = a0[v] + bv;
      out[(size_t)(r0 + 16) * HID + hOut] = a1[v] + bv;
    }
  }
}

// ---------------------------------------------------------------------------
extern "C" void kernel_launch(void* const* d_in, const int* in_sizes, int n_in,
                              void* d_out, int out_size, void* d_ws, size_t ws_size,
                              hipStream_t stream) {
  const float* x  = (const float*)d_in[0];  // (2,1024,768)
  const float* pw = (const float*)d_in[1];  // (768, 64*768)
  const float* pb = (const float*)d_in[2];  // (768,)
  float* out = (float*)d_out;               // (2,1024,768)

  char* wsb = (char*)d_ws;
  __bf16* Xc = (__bf16*)(wsb);
  __bf16* Xs = (__bf16*)(wsb + (size_t)ROWS * HID * 2);
  __bf16* Qc = (__bf16*)(wsb + (size_t)ROWS * HID * 4);
  __bf16* Qs = (__bf16*)(wsb + (size_t)ROWS * HID * 4 + (size_t)WIN * HID * HID * 2);

  dim3 g1(ROWS / 16, HID / 64);
  k_xtransform<<<g1, 128, 0, stream>>>(x, Xc, Xs);

  k_qbuild<<<(HID * HID) / 256, 256, 0, stream>>>(pw, Qc, Qs);

  dim3 g3(ROWS / 128, HID / 64);
  k_main<<<g3, 256, 0, stream>>>(Xc, Xs, Qc, Qs, pb, out);
}
